// MultiHeadAttention_39865886441744
// MI455X (gfx1250) — matrix operand order repaired and checked by reading, not compiled, not run on hardware
//
#include <hip/hip_runtime.h>
#include <hip/hip_bf16.h>

typedef __bf16 bf16;
typedef __attribute__((ext_vector_type(16))) __bf16 v16bf;
typedef __attribute__((ext_vector_type(8)))  float  v8f;

#define CB 4
#define CS 2048
#define CD 1024
#define CH 16
#define CHD 64
#define CSCALE 0.125f   // 64^-0.5

// ---------------------------------------------------------------------------
// Fragment gathers (wave32, per CDNA5 ISA §7.12.2)
// A 16x32 bf16: lane<16 -> row=lane, K={0..7,16..23}; lane>=16 -> row=lane-16,
//               K={8..15,24..31}.  (VGPR j holds K=2j,2j+1)
// B 32x16 bf16: lanes 0-15 hold N=lane, K={0..7,16..23}; lanes 16-31 hold N=lane-16,
//               K={8..15,24..31} (same k order as A).
// C/D 16x16 f32: VGPR r -> M = 8*(lane/16)+r, N = lane%16.
// ---------------------------------------------------------------------------
__device__ __forceinline__ v16bf frag_a(const bf16* base, int ld, int lane) {
    const int half = lane >> 4, row = lane & 15;
    const bf16* p = base + row * ld;
    v16bf f;
    uint4* fv = (uint4*)&f;
    fv[0] = *(const uint4*)(p + half * 8);        // K = half*8 .. +7
    fv[1] = *(const uint4*)(p + 16 + half * 8);   // K = 16+half*8 .. +7
    return f;
}

// B(k,n) stored N-major in LDS: lds[n*ld + k] (contiguous in k)
__device__ __forceinline__ v16bf frag_b_nmajor(const bf16* base, int ld, int lane) {
    const int col = lane & 15, kb = (lane >> 4) * 8;
    const bf16* p = base + col * ld + kb;
    v16bf f;
    uint4* fv = (uint4*)&f;
    fv[0] = *(const uint4*)(p);
    fv[1] = *(const uint4*)(p + 16);
    return f;
}

#define WMMA_BF16(a, b, c) \
    __builtin_amdgcn_wmma_f32_16x16x32_bf16(false, (a), false, (b), (short)0, (c), false, false)

// ---------------------------------------------------------------------------
// fp32 -> bf16 conversion (weights), n multiple of 4
// ---------------------------------------------------------------------------
__global__ __launch_bounds__(256) void cvt_f32_bf16(const float* __restrict__ x,
                                                    bf16* __restrict__ y, int n) {
    int i4 = (blockIdx.x * 256 + threadIdx.x) * 4;
    if (i4 < n) {
        float4 v = *(const float4*)(x + i4);
        y[i4 + 0] = (bf16)v.x;
        y[i4 + 1] = (bf16)v.y;
        y[i4 + 2] = (bf16)v.z;
        y[i4 + 3] = (bf16)v.w;
    }
}

// ---------------------------------------------------------------------------
// GEMM: out[m][n] = sum_k A[m][k] * W[n][k] + bias[n]
// A: f32 or bf16 [M,K]; W: bf16 [N,K]; out: bf16 or f32 [M,N]
// 128 threads = 4 wave32s; 64x64 tile per block; K staged in chunks of 64
// (two 32-deep WMMA sub-chunks per barrier -> 8 WMMA per wave per stage).
// ---------------------------------------------------------------------------
template <typename AT, bool OUT_BF16>
__global__ __launch_bounds__(128) void gemm_xwT(const AT* __restrict__ A,
                                                const bf16* __restrict__ W,
                                                const float* __restrict__ bias,
                                                void* __restrict__ outP,
                                                int M, int N, int K) {
    constexpr int LD = 72;  // 144B stride: 16B aligned, bank-skewed
    __shared__ bf16 sA[64 * LD];
    __shared__ bf16 sW[64 * LD];

    const int tid = threadIdx.x;
    const int lane = tid & 31, wv = tid >> 5;
    const int mblk = blockIdx.y * 64, nblk = blockIdx.x * 64;
    const int mbase = (wv & 1) * 32, nbase = (wv >> 1) * 32;

    v8f acc[2][2] = {};

    for (int kc = 0; kc < K; kc += 64) {
        // stage A tile (64 x 64), converting to bf16 if needed
        if constexpr (sizeof(AT) == 4) {
#pragma unroll
            for (int i = 0; i < 8; ++i) {
                int e = tid + i * 128;           // 1024 float4
                int row = e >> 4, kk = (e & 15) * 4;
                const float* src = (const float*)A + (size_t)(mblk + row) * K + kc + kk;
                float4 v = *(const float4*)src;
                if (kc + 64 < K) __builtin_prefetch(src + 64, 0, 1);
                bf16* d = &sA[row * LD + kk];
                d[0] = (bf16)v.x; d[1] = (bf16)v.y; d[2] = (bf16)v.z; d[3] = (bf16)v.w;
            }
        } else {
#pragma unroll
            for (int i = 0; i < 4; ++i) {
                int e = tid + i * 128;           // 512 uint4 (8 bf16 each)
                int row = e >> 3, kk = (e & 7) * 8;
                const bf16* src = (const bf16*)A + (size_t)(mblk + row) * K + kc + kk;
                *(uint4*)&sA[row * LD + kk] = *(const uint4*)src;
                if (kc + 64 < K) __builtin_prefetch(src + 64, 0, 1);
            }
        }
        // stage W tile (64 rows of n, 64 k) — N-major, contiguous in k
#pragma unroll
        for (int i = 0; i < 4; ++i) {
            int e = tid + i * 128;
            int row = e >> 3, kk = (e & 7) * 8;
            const bf16* src = W + (size_t)(nblk + row) * K + kc + kk;
            *(uint4*)&sW[row * LD + kk] = *(const uint4*)src;
            if (kc + 64 < K) __builtin_prefetch(src + 64, 0, 1);
        }
        __syncthreads();

#pragma unroll
        for (int kc2 = 0; kc2 < 2; ++kc2) {
            v16bf a0 = frag_a(&sA[(mbase + 0) * LD + kc2 * 32], LD, lane);
            v16bf a1 = frag_a(&sA[(mbase + 16) * LD + kc2 * 32], LD, lane);
#pragma unroll
            for (int ni = 0; ni < 2; ++ni) {
                v16bf bb = frag_b_nmajor(&sW[(nbase + ni * 16) * LD + kc2 * 32], LD, lane);
                acc[0][ni] = WMMA_BF16(a0, bb, acc[0][ni]);
                acc[1][ni] = WMMA_BF16(a1, bb, acc[1][ni]);
            }
        }
        __syncthreads();
    }

    const int half = lane >> 4, col16 = lane & 15;
#pragma unroll
    for (int mi = 0; mi < 2; ++mi)
#pragma unroll
        for (int ni = 0; ni < 2; ++ni)
#pragma unroll
            for (int r = 0; r < 8; ++r) {
                int row = mblk + mbase + mi * 16 + half * 8 + r;
                int col = nblk + nbase + ni * 16 + col16;
                float v = acc[mi][ni][r] + bias[col];
                if constexpr (OUT_BF16)
                    ((bf16*)outP)[(size_t)row * N + col] = (bf16)v;
                else
                    ((float*)outP)[(size_t)row * N + col] = v;
            }
}

// ---------------------------------------------------------------------------
// Flash attention: block = (64-query tile, head, batch); 128 threads = 4 waves.
// Wave w owns query rows 16w..16w+15 and their persistent 16x64 ctx accum.
// K/V streamed in 64-key tiles; online softmax with row-owner threads tid<64.
// V is staged TRANSPOSED (Vt[d][key]) so the ctx-GEMM B operand is N-major
// and fragment gathers are contiguous ds_load_b128.
// ---------------------------------------------------------------------------
__global__ __launch_bounds__(128) void attention_kernel(const bf16* __restrict__ Q,
                                                        const bf16* __restrict__ Kx,
                                                        const bf16* __restrict__ V,
                                                        bf16* __restrict__ Ctx) {
    constexpr int LD = 72;  // 144B stride: 16B aligned, bank-skewed
    __shared__ bf16 Qs[64 * LD];
    __shared__ bf16 Ks[64 * LD];
    __shared__ bf16 Vt[64 * LD];     // transposed: Vt[d][key]
    __shared__ bf16 Pb[64 * LD];
    __shared__ float Sf[64 * 65];
    __shared__ float alphaS[64];
    __shared__ float linvS[64];

    const int tid = threadIdx.x;
    const int lane = tid & 31, wv = tid >> 5;
    const int half = lane >> 4, col16 = lane & 15;
    const int qt = blockIdx.x, h = blockIdx.y, b = blockIdx.z;
    const int rbase = wv * 16;

    // load Q tile once: 64 rows x 64 bf16 (head slice of [B,S,D])
    const bf16* Qbase = Q + ((size_t)b * CS + qt * 64) * CD + h * CHD;
#pragma unroll
    for (int i = 0; i < 4; ++i) {
        int e = tid + i * 128;               // 512 uint4
        int row = e >> 3, c = (e & 7) * 8;
        *(uint4*)&Qs[row * LD + c] = *(const uint4*)(Qbase + (size_t)row * CD + c);
    }

    float m_i = -3.0e38f, l_i = 0.0f;        // row state (tid<64 owns row tid)
    v8f o[4] = {};                            // 16x64 ctx strip per wave

    for (int kt = 0; kt < CS; kt += 64) {
        const bf16* Kb = Kx + ((size_t)b * CS + kt) * CD + h * CHD;
        const bf16* Vb = V + ((size_t)b * CS + kt) * CD + h * CHD;
#pragma unroll
        for (int i = 0; i < 4; ++i) {
            int e = tid + i * 128;
            int row = e >> 3, c = (e & 7) * 8;
            // K tile: natural [key][d] layout (N-major for the score GEMM)
            *(uint4*)&Ks[row * LD + c] = *(const uint4*)(Kb + (size_t)row * CD + c);
            // V tile: transpose on the fly -> Vt[d][key]
            uint4 vv = *(const uint4*)(Vb + (size_t)row * CD + c);
            const bf16* t = (const bf16*)&vv;
#pragma unroll
            for (int j = 0; j < 8; ++j) Vt[(c + j) * LD + row] = t[j];
            // prefetch next key tile while this one computes
            if (kt + 64 < CS) {
                __builtin_prefetch(Kb + 64 * CD + (size_t)row * CD + c, 0, 1);
                __builtin_prefetch(Vb + 64 * CD + (size_t)row * CD + c, 0, 1);
            }
        }
        __syncthreads();

        // S = Q Kᵀ : A = Q strip (16 x 64), B(d,key) = K[key][d] (N-major in Ks)
        v8f sacc[4] = {};
#pragma unroll
        for (int kc = 0; kc < 2; ++kc) {
            v16bf a = frag_a(&Qs[rbase * LD + kc * 32], LD, lane);
#pragma unroll
            for (int nt = 0; nt < 4; ++nt) {
                v16bf bb = frag_b_nmajor(&Ks[(nt * 16) * LD + kc * 32], LD, lane);
                sacc[nt] = WMMA_BF16(a, bb, sacc[nt]);
            }
        }
#pragma unroll
        for (int nt = 0; nt < 4; ++nt)
#pragma unroll
            for (int r = 0; r < 8; ++r)
                Sf[(rbase + half * 8 + r) * 65 + nt * 16 + col16] = sacc[nt][r] * CSCALE;
        __syncthreads();

        // online softmax (one owner thread per query row)
        if (tid < 64) {
            const float* srow = &Sf[tid * 65];
            float cmax = srow[0];
#pragma unroll 8
            for (int c = 1; c < 64; ++c) cmax = fmaxf(cmax, srow[c]);
            float nm = fmaxf(m_i, cmax);
            float alpha = __expf(m_i - nm);
            float ssum = 0.0f;
            bf16* prow = &Pb[tid * LD];
#pragma unroll 8
            for (int c = 0; c < 64; ++c) {
                float p = __expf(srow[c] - nm);
                prow[c] = (bf16)p;
                ssum += p;
            }
            l_i = l_i * alpha + ssum;
            m_i = nm;
            alphaS[tid] = alpha;
        }
        __syncthreads();

        // rescale running ctx, then ctx += P V :
        // A = P strip (16 x 64 keys), B(key,d) = Vt[d][key] (N-major in Vt)
#pragma unroll
        for (int r = 0; r < 8; ++r) {
            float al = alphaS[rbase + half * 8 + r];
#pragma unroll
            for (int nt = 0; nt < 4; ++nt) o[nt][r] *= al;
        }
#pragma unroll
        for (int kc = 0; kc < 2; ++kc) {
            v16bf a = frag_a(&Pb[rbase * LD + kc * 32], LD, lane);
#pragma unroll
            for (int nt = 0; nt < 4; ++nt) {
                v16bf bb = frag_b_nmajor(&Vt[(nt * 16) * LD + kc * 32], LD, lane);
                o[nt] = WMMA_BF16(a, bb, o[nt]);
            }
        }
        __syncthreads();
    }

    if (tid < 64) linvS[tid] = 1.0f / l_i;
    __syncthreads();

    bf16* Cb = Ctx + ((size_t)b * CS + qt * 64) * CD + h * CHD;
#pragma unroll
    for (int nt = 0; nt < 4; ++nt)
#pragma unroll
        for (int r = 0; r < 8; ++r) {
            int row = rbase + half * 8 + r;
            int col = nt * 16 + col16;
            Cb[(size_t)row * CD + col] = (bf16)(o[nt][r] * linvS[row]);
        }
}

// ---------------------------------------------------------------------------
extern "C" void kernel_launch(void* const* d_in, const int* in_sizes, int n_in,
                              void* d_out, int out_size, void* d_ws, size_t ws_size,
                              hipStream_t stream) {
    const float* query = (const float*)d_in[0];
    const float* key   = (const float*)d_in[1];
    const float* value = (const float*)d_in[2];
    const float* Wq = (const float*)d_in[3];  const float* bq = (const float*)d_in[4];
    const float* Wk = (const float*)d_in[5];  const float* bk = (const float*)d_in[6];
    const float* Wv = (const float*)d_in[7];  const float* bv = (const float*)d_in[8];
    const float* Wo = (const float*)d_in[9];  const float* bo = (const float*)d_in[10];
    float* out = (float*)d_out;

    const int M = CB * CS;        // 8192
    const int N = CD, K = CD;     // 1024
    const size_t WBYTES = (size_t)CD * CD * sizeof(bf16);       // 2 MB
    const size_t ABYTES = (size_t)M * CD * sizeof(bf16);        // 16 MB

    char* ws = (char*)d_ws;
    bf16* Wqb = (bf16*)(ws);
    bf16* Wkb = (bf16*)(ws + WBYTES);
    bf16* Wvb = (bf16*)(ws + 2 * WBYTES);
    bf16* Wob = (bf16*)(ws + 3 * WBYTES);
    bf16* Qb  = (bf16*)(ws + 4 * WBYTES);
    bf16* Kb  = (bf16*)(ws + 4 * WBYTES + ABYTES);
    bf16* Vb  = (bf16*)(ws + 4 * WBYTES + 2 * ABYTES);
    bf16* Cxb = (bf16*)(ws + 4 * WBYTES + 3 * ABYTES);

    // 1) weights -> bf16
    cvt_f32_bf16<<<1024, 256, 0, stream>>>(Wq, Wqb, CD * CD);
    cvt_f32_bf16<<<1024, 256, 0, stream>>>(Wk, Wkb, CD * CD);
    cvt_f32_bf16<<<1024, 256, 0, stream>>>(Wv, Wvb, CD * CD);
    cvt_f32_bf16<<<1024, 256, 0, stream>>>(Wo, Wob, CD * CD);

    // 2) Q/K/V projections (f32 activations in, bf16 out, bias fused)
    dim3 ggrid(N / 64, M / 64);
    gemm_xwT<float, true><<<ggrid, 128, 0, stream>>>(query, Wqb, bq, Qb, M, N, K);
    gemm_xwT<float, true><<<ggrid, 128, 0, stream>>>(key,   Wkb, bk, Kb, M, N, K);
    gemm_xwT<float, true><<<ggrid, 128, 0, stream>>>(value, Wvb, bv, Vb, M, N, K);

    // 3) attention
    attention_kernel<<<dim3(CS / 64, CH, CB), 128, 0, stream>>>(Qb, Kb, Vb, Cxb);

    // 4) output projection (bf16 in, f32 out, bias fused)
    gemm_xwT<bf16, false><<<ggrid, 128, 0, stream>>>(Cxb, Wob, bo, out, M, N, K);
}
